// SNNController_45062796869844
// MI455X (gfx1250) — matrix operand order repaired and checked by reading, not compiled, not run on hardware
//
#include <hip/hip_runtime.h>

typedef float    v2f  __attribute__((ext_vector_type(2)));
typedef float    v8f  __attribute__((ext_vector_type(8)));
typedef _Float16 v16h __attribute__((ext_vector_type(16)));

#define T_STEPS 1000
#define BATCH   8192
#define IN_F    9
#define HID     96
#define OUT_F   3
#define NTILES  6            // HID / 16
#define BETA    0.92f
#define THRESH  1.0f

union H16 { unsigned u[8]; v16h h; };   // 8 dwords = 16 f16 lanes

__global__ __launch_bounds__(128, 1)
void SNNController_45062796869844_kernel(
    const float* __restrict__ x,    // (T, B, I)
    const float* __restrict__ W1,   // (H, I)
    const float* __restrict__ b1,   // (H)
    const float* __restrict__ W2,   // (O, H)
    const float* __restrict__ b2,   // (O)
    float* __restrict__ spk_out,    // (T, B, O)
    float* __restrict__ mem_out)    // (T, B, O)
{
    const int lane   = threadIdx.x & 31;
    const int n      = lane & 15;        // batch row within tile (N dim of layer1')
    const int hiHalf = lane >> 4;
    const int wave   = (int)((blockIdx.x * blockDim.x + threadIdx.x) >> 5);
    const int base   = wave * 16;        // batch tile base
    if (base >= BATCH) return;

    // ---- Layer-1 A fragments: W1 tiles (16h x 12k), bias folded in at K=9 ----
    // f32 A 16x4 layout: lane&15 = M(h-local), K = 4*ks + 2*hiHalf + vgpr
    v2f w1f[NTILES][3];
#pragma unroll
    for (int nt = 0; nt < NTILES; ++nt) {
        const int h = nt * 16 + n;
        {   // ks = 0: K = {0,1} / {2,3}
            const int kb = 2 * hiHalf;
            v2f f; f.x = W1[h * IN_F + kb]; f.y = W1[h * IN_F + kb + 1];
            w1f[nt][0] = f;
        }
        {   // ks = 1: K = {4,5} / {6,7}
            const int kb = 4 + 2 * hiHalf;
            v2f f; f.x = W1[h * IN_F + kb]; f.y = W1[h * IN_F + kb + 1];
            w1f[nt][1] = f;
        }
        {   // ks = 2: K = {8, 9=bias} / {10,11} = zero pad
            v2f f;
            f.x = hiHalf ? 0.0f : W1[h * IN_F + 8];
            f.y = hiHalf ? 0.0f : b1[h];
            w1f[nt][2] = f;
        }
    }

    // ---- Layer-2 B fragments: W2^T (96k x 16o) as f16, 3 chunks of K=32 ----
    // f16 B 32x16 layout: lane&15 = N(o), dword v holds K = 32c + 8*hiHalf + 2v + (v>=4 ? 8 : 0), +1
    H16 w2b[3];
#pragma unroll
    for (int c = 0; c < 3; ++c) {
#pragma unroll
        for (int v = 0; v < 8; ++v) {
            const int k = 32 * c + 8 * hiHalf + 2 * v + (v >= 4 ? 8 : 0);
            float wa = (n < OUT_F) ? W2[n * HID + k]     : 0.0f;
            float wb = (n < OUT_F) ? W2[n * HID + k + 1] : 0.0f;
            w2b[c].h[2 * v]     = (_Float16)wa;
            w2b[c].h[2 * v + 1] = (_Float16)wb;
        }
    }

    // per-lane bias constants for layer-2 seed: b2 and b2 - THRESH (spiked case)
    const float b2f  = (n == 0) ? b2[0] : (n == 1) ? b2[1] : b2[2];
    const float b2fm = b2f - THRESH;

    // ---- Persistent state ----
    v8f mem1[NTILES];        // (96h x 16m) tiles, transposed C/D layout
#pragma unroll
    for (int nt = 0; nt < NTILES; ++nt)
#pragma unroll
        for (int r = 0; r < 8; ++r) mem1[nt][r] = 0.0f;
    v8f mem2;                // rows m = r + 8*hiHalf, this lane's o = n
#pragma unroll
    for (int r = 0; r < 8; ++r) mem2[r] = 0.0f;

    const int kb0 = 2 * hiHalf;
    const float* xbase   = x + ((size_t)base + n) * IN_F;
    const size_t tstride = (size_t)BATCH * IN_F;

    // software-pipelined x fragment scalars (current timestep)
    float cx0 = xbase[kb0], cx1 = xbase[kb0 + 1];
    float cx4 = xbase[4 + kb0], cx5 = xbase[4 + kb0 + 1];
    float cx8 = xbase[8];    // loaded by all lanes; masked into fragment below

    for (int t = 0; t < T_STEPS; ++t) {
        // ---- issue loads for t+1 early (clamped on last step) ----
        const float* xn = xbase + (size_t)((t + 1 < T_STEPS) ? t + 1 : t) * tstride;
        __builtin_prefetch(xn + tstride, 0, 0);   // speculative t+2 prefetch
        float nx0 = xn[kb0], nx1 = xn[kb0 + 1];
        float nx4 = xn[4 + kb0], nx5 = xn[4 + kb0 + 1];
        float nx8 = xn[8];

        // ---- Layer-1 B fragments: x^T (12k x 16m), constant-1 at K=9 ----
        v2f xf0; xf0.x = cx0; xf0.y = cx1;
        v2f xf1; xf1.x = cx4; xf1.y = cx5;
        v2f xf2; xf2.x = hiHalf ? 0.0f : cx8;
        xf2.y = hiHalf ? 0.0f : 1.0f;

        // ---- Layer 1: seed C = BETA*mem1 - rst(old), WMMA chain yields mem1_new ----
        H16 packA[3];   // layer-2 A fragments, K-chunks of 32
#pragma unroll
        for (int nt = 0; nt < NTILES; ++nt) {
            v8f c0;
#pragma unroll
            for (int e = 0; e < 8; ++e) {
                float m1 = mem1[nt][e];
                float rs = (m1 > THRESH) ? THRESH : 0.0f;   // reset from OLD mem
                c0[e] = BETA * m1 - rs;
            }
            v8f acc;
            acc = __builtin_amdgcn_wmma_f32_16x16x4_f32(
                false, w1f[nt][0], false, xf0, (short)0, c0,  false, false);
            acc = __builtin_amdgcn_wmma_f32_16x16x4_f32(
                false, w1f[nt][1], false, xf1, (short)0, acc, false, false);
            acc = __builtin_amdgcn_wmma_f32_16x16x4_f32(
                false, w1f[nt][2], false, xf2, (short)0, acc, false, false);
            mem1[nt] = acc;   // D is the new membrane state (bias via K=9 column)
#pragma unroll
            for (int rp = 0; rp < 4; ++rp) {
                // spike pair -> packed f16 {0,1}: dword v of chunk nt>>1
                unsigned d = (acc[2 * rp]     > THRESH ? 0x00003C00u : 0u)
                           | (acc[2 * rp + 1] > THRESH ? 0x3C000000u : 0u);
                packA[nt >> 1].u[(nt & 1) * 4 + rp] = d;
            }
        }

        // ---- Layer 2: seed C = BETA*mem2 + (spiked ? b2-1 : b2), f16 WMMA K=96 ----
        v8f c2;
#pragma unroll
        for (int r = 0; r < 8; ++r) {
            float m2  = mem2[r];
            float sel = (m2 > THRESH) ? b2fm : b2f;
            c2[r] = BETA * m2 + sel;
        }
        v8f d2;
        d2 = __builtin_amdgcn_wmma_f32_16x16x32_f16(
            false, packA[0].h, false, w2b[0].h, (short)0, c2, false, false);
        d2 = __builtin_amdgcn_wmma_f32_16x16x32_f16(
            false, packA[1].h, false, w2b[1].h, (short)0, d2, false, false);
        d2 = __builtin_amdgcn_wmma_f32_16x16x32_f16(
            false, packA[2].h, false, w2b[2].h, (short)0, d2, false, false);
        mem2 = d2;   // D is the new layer-2 membrane state (bias folded into seed)

        // ---- outputs: row m = r + 8*hiHalf, col o = n (<3) ----
        const size_t ob = ((size_t)t * BATCH + base + 8 * hiHalf) * OUT_F + (size_t)n;
#pragma unroll
        for (int r = 0; r < 8; ++r) {
            float m2 = d2[r];
            if (n < OUT_F) {
                const size_t idx = ob + (size_t)r * OUT_F;
                mem_out[idx] = m2;
                spk_out[idx] = (m2 > THRESH) ? 1.0f : 0.0f;
            }
        }

        // rotate pipelined x registers
        cx0 = nx0; cx1 = nx1; cx4 = nx4; cx5 = nx5; cx8 = nx8;
    }
}

extern "C" void kernel_launch(void* const* d_in, const int* in_sizes, int n_in,
                              void* d_out, int out_size, void* d_ws, size_t ws_size,
                              hipStream_t stream) {
    (void)in_sizes; (void)n_in; (void)out_size; (void)d_ws; (void)ws_size;
    const float* x  = (const float*)d_in[0];
    const float* W1 = (const float*)d_in[1];
    const float* b1 = (const float*)d_in[2];
    const float* W2 = (const float*)d_in[3];
    const float* b2 = (const float*)d_in[4];
    float* out = (float*)d_out;
    float* spk = out;                                             // (T,B,O)
    float* mem = out + (size_t)T_STEPS * BATCH * OUT_F;           // (T,B,O)

    const int tiles   = BATCH / 16;          // 512 waves, one M=16 tile each
    const int threads = 128;                 // 4 waves per block
    const int blocks  = (tiles * 32 + threads - 1) / threads;     // 128 blocks
    SNNController_45062796869844_kernel<<<blocks, threads, 0, stream>>>(
        x, W1, b1, W2, b2, spk, mem);
}
